// Transformer_89704686944458
// MI455X (gfx1250) — matrix operand order repaired and checked by reading, not compiled, not run on hardware
//
#include <hip/hip_runtime.h>
#include <hip/hip_bf16.h>
#include <math.h>

// ---------------- types ----------------
typedef __attribute__((ext_vector_type(16))) __bf16 v16bf;
typedef __attribute__((ext_vector_type(8)))  __bf16 bf16x8;
typedef __attribute__((ext_vector_type(2)))  __bf16 bf16x2;
typedef __attribute__((ext_vector_type(8)))  float  v8f;
typedef __attribute__((ext_vector_type(4)))  int    int4v;

// async global->LDS copy support (gfx1250), compile-guarded
#if defined(__has_builtin)
#if __has_builtin(__builtin_amdgcn_global_load_async_to_lds_b128)
#define USE_ASYNC_LDS 1
#endif
#endif
#ifndef USE_ASYNC_LDS
#define USE_ASYNC_LDS 0
#endif

#if USE_ASYNC_LDS
typedef __attribute__((address_space(1))) int4v glb_i4_t;   // global int4 ptr
typedef __attribute__((address_space(3))) int4v lds_i4_t;   // LDS int4 ptr
#if __has_builtin(__builtin_amdgcn_s_wait_asynccnt)
#define WAIT_ASYNC() __builtin_amdgcn_s_wait_asynccnt(0)
#else
#define WAIT_ASYNC() asm volatile("s_wait_asynccnt 0" ::: "memory")
#endif
#endif

// ---------------- problem constants ----------------
#define BATCH 4
#define SEQ   4096
#define DM    768
#define NH    12
#define DHD   64
#define HID   3072
#define NDEPTH 4
#define ROWS  (BATCH*SEQ)       // 16384
#define LDT   40                // padded LDS row stride (bf16), 80B rows

// =====================================================================
// Tiled bf16 WMMA GEMM: C(MxN) = A(MxK,bf16) * W(KxN,f32->bf16) [+ epilogue]
// MODE 0: outF = C ; MODE 1: outF = C+bias+resid ; MODE 2: outB = bf16(gelu(C+bias))
// 256 thr = 8 waves; block tile 128x128; wave tile 32x64 (2x4 wmma);
// K-chunks of 32, LDS double-buffered, one barrier per chunk;
// A tile staged via async global->LDS, B converted f32->bf16 (packed b32 stores).
// =====================================================================
template<int MODE>
__global__ __launch_bounds__(256) void gemm_bf16_wmma(
    const __bf16* __restrict__ A, const float* __restrict__ W,
    const float* __restrict__ bias, const float* resid,
    float* outF, __bf16* outB, int M, int N, int K)
{
    __shared__ __bf16 As[2][128 * LDT];
    __shared__ __bf16 Bs[2][128 * LDT];

    const int tid = threadIdx.x;
    const unsigned blockM = blockIdx.y * 128u;
    const unsigned blockN = blockIdx.x * 128u;
    const int lane = tid & 31, wv = tid >> 5;
    const int wm = wv & 3;      // wave M group (0..3) -> 32 rows each
    const int wn = wv >> 2;     // wave N group (0..1) -> 64 cols each
    const int lrow = lane & 15, lh = lane >> 4;

    v8f acc[2][4];
#pragma unroll
    for (int mi = 0; mi < 2; ++mi)
#pragma unroll
        for (int ni = 0; ni < 4; ++ni)
#pragma unroll
            for (int r = 0; r < 8; ++r) acc[mi][ni][r] = 0.0f;

    // cooperative-load assignments
    const int aM  = tid >> 1;            // 0..127
    const int aK  = (tid & 1) * 16;      // 0 / 16
    const int bK2 = (tid >> 4) * 2;      // 0,2,..,30  (two adjacent k rows)
    const int bN8 = (tid & 15) * 8;      // 0,8,..,120 (eight n columns)

    const unsigned aBase = (blockM + (unsigned)aM) * (unsigned)K + (unsigned)aK;
    const unsigned wBase = (unsigned)bK2 * (unsigned)N + blockN + (unsigned)bN8;

    bf16x8 a0, a1;                       // A staging regs (sync fallback)
    float4 wa0, wa1, wb0, wb1;           // B staging regs: rows k, k+1

    auto load_phase = [&](int kc) {
        const unsigned k0 = (unsigned)kc * 32u;
        const float* wp = W + wBase + k0 * (unsigned)N;
        wa0 = *(const float4*)(wp);
        wa1 = *(const float4*)(wp + 4);
        wb0 = *(const float4*)(wp + (unsigned)N);
        wb1 = *(const float4*)(wp + (unsigned)N + 4);
        const __bf16* gp = A + aBase + k0;
#if USE_ASYNC_LDS
        __bf16* lp = &As[kc & 1][aM * LDT + aK];
        __builtin_amdgcn_global_load_async_to_lds_b128(
            (glb_i4_t*)gp, (lds_i4_t*)lp, 0, 0);
        __builtin_amdgcn_global_load_async_to_lds_b128(
            (glb_i4_t*)gp, (lds_i4_t*)lp, 16, 0);
#else
        a0 = *(const bf16x8*)(gp);
        a1 = *(const bf16x8*)(gp + 8);
#endif
    };

    auto store_phase = [&](int kc) {
        const int buf = kc & 1;
#if !USE_ASYNC_LDS
        *(bf16x8*)&As[buf][aM * LDT + aK]     = a0;
        *(bf16x8*)&As[buf][aM * LDT + aK + 8] = a1;
#endif
        // Bs[n][k]: pack bf16(W[k][n]), bf16(W[k+1][n]) -> one b32 store
        __bf16* bp = &Bs[buf][bN8 * LDT + bK2];
        auto pk = [](float lo, float hi) {
            bf16x2 p; p[0] = (__bf16)lo; p[1] = (__bf16)hi; return p;
        };
        *(bf16x2*)(bp + 0 * LDT) = pk(wa0.x, wb0.x);
        *(bf16x2*)(bp + 1 * LDT) = pk(wa0.y, wb0.y);
        *(bf16x2*)(bp + 2 * LDT) = pk(wa0.z, wb0.z);
        *(bf16x2*)(bp + 3 * LDT) = pk(wa0.w, wb0.w);
        *(bf16x2*)(bp + 4 * LDT) = pk(wa1.x, wb1.x);
        *(bf16x2*)(bp + 5 * LDT) = pk(wa1.y, wb1.y);
        *(bf16x2*)(bp + 6 * LDT) = pk(wa1.z, wb1.z);
        *(bf16x2*)(bp + 7 * LDT) = pk(wa1.w, wb1.w);
    };

    auto compute = [&](int buf) {
        v16bf afrag[2], bfrag[4];
#pragma unroll
        for (int mi = 0; mi < 2; ++mi) {
            // A lane layout: M = lane%16; j<8: K = lh*8+j ; j>=8: K = 16+lh*8+(j-8)
            const __bf16* ap = &As[buf][(wm * 32 + mi * 16 + lrow) * LDT + lh * 8];
            bf16x8 lo = *(const bf16x8*)ap;
            bf16x8 hi = *(const bf16x8*)(ap + 16);
#pragma unroll
            for (int j = 0; j < 8; ++j) { afrag[mi][j] = lo[j]; afrag[mi][j + 8] = hi[j]; }
        }
#pragma unroll
        for (int ni = 0; ni < 4; ++ni) {
            // B lane layout: N = lane%16; j<8: K = lh*8+j ; j>=8: K = 16+lh*8+(j-8)
            const __bf16* bp = &Bs[buf][(wn * 64 + ni * 16 + lrow) * LDT + lh * 8];
            bf16x8 lo = *(const bf16x8*)bp;
            bf16x8 hi = *(const bf16x8*)(bp + 16);
#pragma unroll
            for (int j = 0; j < 8; ++j) { bfrag[ni][j] = lo[j]; bfrag[ni][j + 8] = hi[j]; }
        }
#pragma unroll
        for (int mi = 0; mi < 2; ++mi)
#pragma unroll
            for (int ni = 0; ni < 4; ++ni)
                acc[mi][ni] = __builtin_amdgcn_wmma_f32_16x16x32_bf16(
                    false, afrag[mi], false, bfrag[ni],
                    (short)0, acc[mi][ni], false, false);
    };

    const int nk = K >> 5;
    load_phase(0);
    store_phase(0);
#if USE_ASYNC_LDS
    WAIT_ASYNC();
#endif
    __syncthreads();

    for (int k = 0; k < nk; ++k) {
        const bool more = (k + 1) < nk;
        if (more) load_phase(k + 1);
        compute(k & 1);
        if (more) store_phase(k + 1);
#if USE_ASYNC_LDS
        WAIT_ASYNC();
#endif
        __syncthreads();
    }

    // ---- epilogue; C layout: VGPR r -> M = r + 8*lh, N = lane%16 ----
#pragma unroll
    for (int mi = 0; mi < 2; ++mi) {
#pragma unroll
        for (int ni = 0; ni < 4; ++ni) {
            const unsigned colg = blockN + (unsigned)(wn * 64 + ni * 16 + lrow);
            const unsigned rowb = blockM + (unsigned)(wm * 32 + mi * 16 + lh * 8);
            const unsigned off0 = rowb * (unsigned)N + colg;
            v8f cc = acc[mi][ni];
            const float bcol = (MODE != 0) ? bias[colg] : 0.0f;
#pragma unroll
            for (int r = 0; r < 8; ++r) {
                const unsigned o = off0 + (unsigned)r * (unsigned)N;
                if (MODE == 0) {
                    outF[o] = cc[r];
                } else if (MODE == 1) {
                    outF[o] = cc[r] + bcol + resid[o];
                } else {
                    float t = cc[r] + bcol;
                    outB[o] = (__bf16)(0.5f * t * (1.0f + erff(t * 0.70710678118654752f)));
                }
            }
        }
    }
}

// =====================================================================
// LayerNorm over last dim (768) -> bf16
// =====================================================================
__global__ __launch_bounds__(256) void ln_kernel(
    const float* __restrict__ x, const float* __restrict__ g,
    const float* __restrict__ b, __bf16* __restrict__ out)
{
    __shared__ float red[256];
    const unsigned row = blockIdx.x;
    const int tid = threadIdx.x;
    const float* xr = x + row * (unsigned)DM;
    float v0 = xr[tid], v1 = xr[tid + 256], v2 = xr[tid + 512];

    red[tid] = v0 + v1 + v2;
    __syncthreads();
    for (int s = 128; s > 0; s >>= 1) { if (tid < s) red[tid] += red[tid + s]; __syncthreads(); }
    float mu = red[0] * (1.0f / DM);
    __syncthreads();

    float d0 = v0 - mu, d1 = v1 - mu, d2 = v2 - mu;
    red[tid] = d0 * d0 + d1 * d1 + d2 * d2;
    __syncthreads();
    for (int s = 128; s > 0; s >>= 1) { if (tid < s) red[tid] += red[tid + s]; __syncthreads(); }
    float rs = rsqrtf(red[0] * (1.0f / DM) + 1e-5f);

    __bf16* o = out + row * (unsigned)DM;
    o[tid]       = (__bf16)(d0 * rs * g[tid]       + b[tid]);
    o[tid + 256] = (__bf16)(d1 * rs * g[tid + 256] + b[tid + 256]);
    o[tid + 512] = (__bf16)(d2 * rs * g[tid + 512] + b[tid + 512]);
}

// =====================================================================
// Attention stage 1: nrm2[b,h,d] = sum_n w[b,n,h,d]^2   (48 blocks)
// =====================================================================
__global__ __launch_bounds__(256) void att_nrm2(
    const float* __restrict__ w, float* __restrict__ nrm2)
{
    __shared__ float red[256];
    const int bh = blockIdx.x, b = bh / NH, h = bh % NH;
    const int tid = threadIdx.x, d = tid & 63, slice = tid >> 6;
    float acc = 0.0f;
    const unsigned cbase = (unsigned)b * SEQ * DM + h * DHD + d;
    for (int n = slice * (SEQ / 4); n < (slice + 1) * (SEQ / 4); ++n) {
        float v = w[cbase + (unsigned)n * DM];
        acc += v * v;
    }
    red[tid] = acc;
    __syncthreads();
    if (tid < 64)
        nrm2[(unsigned)bh * DHD + d] = red[tid] + red[tid + 64] + red[tid + 128] + red[tid + 192];
}

// =====================================================================
// Attention stage 2: logits -> softmax Pi, T = sum(Pi)   (48 blocks)
// =====================================================================
__global__ __launch_bounds__(256) void att_softmax(
    const float* __restrict__ w, const float* __restrict__ nrm2,
    const float* __restrict__ temp, float* __restrict__ Pi,
    float* __restrict__ Tsum)
{
    __shared__ float inv2[DHD];
    __shared__ float red[256];
    const int bh = blockIdx.x, b = bh / NH, h = bh % NH;
    const int tid = threadIdx.x;

    if (tid < DHD) {
        float nr = sqrtf(nrm2[(unsigned)bh * DHD + tid]);
        float m  = fmaxf(nr, 1e-12f);
        inv2[tid] = 1.0f / (m * m);
    }
    __syncthreads();

    const float tp = temp[h];
    float lg[16];
    float lmax = -3.0e38f;
#pragma unroll 1
    for (int j = 0; j < 16; ++j) {
        const int n = tid + j * 256;
        const float4* wr = (const float4*)(w + (unsigned)b * SEQ * DM + (unsigned)n * DM + h * DHD);
        float s = 0.0f;
#pragma unroll
        for (int d4 = 0; d4 < 16; ++d4) {
            float4 v = wr[d4];
            s += v.x * v.x * inv2[d4 * 4 + 0] + v.y * v.y * inv2[d4 * 4 + 1]
               + v.z * v.z * inv2[d4 * 4 + 2] + v.w * v.w * inv2[d4 * 4 + 3];
        }
        lg[j] = s * tp;
        lmax = fmaxf(lmax, lg[j]);
    }
    red[tid] = lmax;
    __syncthreads();
    for (int s = 128; s > 0; s >>= 1) { if (tid < s) red[tid] = fmaxf(red[tid], red[tid + s]); __syncthreads(); }
    const float gmax = red[0];
    __syncthreads();

    float lsum = 0.0f;
#pragma unroll
    for (int j = 0; j < 16; ++j) { lg[j] = expf(lg[j] - gmax); lsum += lg[j]; }
    red[tid] = lsum;
    __syncthreads();
    for (int s = 128; s > 0; s >>= 1) { if (tid < s) red[tid] += red[tid + s]; __syncthreads(); }
    const float S = red[0];
    __syncthreads();

    float t = 0.0f;
    float* Prow = Pi + (unsigned)bh * SEQ;
#pragma unroll
    for (int j = 0; j < 16; ++j) {
        float p = lg[j] / S;
        Prow[tid + j * 256] = p;
        t += p;
    }
    red[tid] = t;
    __syncthreads();
    for (int s = 128; s > 0; s >>= 1) { if (tid < s) red[tid] += red[tid + s]; __syncthreads(); }
    if (tid == 0) Tsum[bh] = red[0];
}

// =====================================================================
// Attention stage 3: dots = sum_n Pi_n * w^2 ; attn = 1/(1+dots)  (48 blocks)
// =====================================================================
__global__ __launch_bounds__(256) void att_dots(
    const float* __restrict__ w, const float* __restrict__ Pi,
    const float* __restrict__ Tsum, float* __restrict__ attn)
{
    __shared__ float red[256];
    const int bh = blockIdx.x, b = bh / NH, h = bh % NH;
    const int tid = threadIdx.x, d = tid & 63, slice = tid >> 6;
    float acc = 0.0f;
    const unsigned cbase = (unsigned)b * SEQ * DM + h * DHD + d;
    const float* Prow = Pi + (unsigned)bh * SEQ;
    for (int n = slice * (SEQ / 4); n < (slice + 1) * (SEQ / 4); ++n) {
        float v = w[cbase + (unsigned)n * DM];
        acc += Prow[n] * v * v;
    }
    red[tid] = acc;
    __syncthreads();
    if (tid < 64) {
        float dots = (red[tid] + red[tid + 64] + red[tid + 128] + red[tid + 192])
                     * (1.0f / (Tsum[bh] + 1e-8f));
        attn[(unsigned)bh * DHD + d] = 1.0f / (1.0f + dots);
    }
}

// =====================================================================
// Attention stage 4: aout = bf16( -(w*Pi)*attn )   (elementwise)
// =====================================================================
__global__ __launch_bounds__(256) void att_out(
    const float* __restrict__ w, const float* __restrict__ Pi,
    const float* __restrict__ attn, __bf16* __restrict__ aout)
{
    const unsigned idx = blockIdx.x * 256u + threadIdx.x;
    const unsigned b = idx / ((unsigned)SEQ * DM);
    const unsigned rem = idx - b * (unsigned)SEQ * DM;
    const unsigned n = rem / DM;
    const unsigned c = rem - n * DM;
    const unsigned h = c / DHD, d = c - h * DHD;
    const unsigned bh = b * NH + h;
    float v = -(w[idx] * Pi[bh * SEQ + n]) * attn[bh * DHD + d];
    aout[idx] = (__bf16)v;
}

// =====================================================================
// residual init: d_out = x
// =====================================================================
__global__ __launch_bounds__(256) void copy_f32(
    const float* __restrict__ in, float* __restrict__ out)
{
    const unsigned i = blockIdx.x * 256u + threadIdx.x;
    out[i] = in[i];
}

// =====================================================================
// host side
// =====================================================================
extern "C" void kernel_launch(void* const* d_in, const int* in_sizes, int n_in,
                              void* d_out, int out_size, void* d_ws, size_t ws_size,
                              hipStream_t stream)
{
    const float* x      = (const float*)d_in[0];
    const float* ln1_g  = (const float*)d_in[1];
    const float* ln1_b  = (const float*)d_in[2];
    const float* qkv_w  = (const float*)d_in[3];
    const float* temp   = (const float*)d_in[4];
    const float* out_w  = (const float*)d_in[5];
    const float* out_b  = (const float*)d_in[6];
    const float* ln2_g  = (const float*)d_in[7];
    const float* ln2_b  = (const float*)d_in[8];
    const float* fc1_w  = (const float*)d_in[9];
    const float* fc1_b  = (const float*)d_in[10];
    const float* fc2_w  = (const float*)d_in[11];
    const float* fc2_b  = (const float*)d_in[12];
    float* xo = (float*)d_out;   // residual stream (fp32)

    // ---- workspace carve-up ----
    char* ws = (char*)d_ws;
    size_t off = 0;
    __bf16* hbf   = (__bf16*)(ws + off); off += (size_t)ROWS * DM * 2;
    float*  wbuf  = (float*) (ws + off); off += (size_t)ROWS * DM * 4;
    __bf16* aout  = (__bf16*)(ws + off); off += (size_t)ROWS * DM * 2;
    __bf16* midbf = (__bf16*)(ws + off); off += (size_t)ROWS * HID * 2;
    float*  nrm2  = (float*) (ws + off); off += (size_t)BATCH * NH * DHD * 4;
    float*  Pi    = (float*) (ws + off); off += (size_t)BATCH * NH * SEQ * 4;
    float*  Tsum  = (float*) (ws + off); off += 256;
    float*  attn  = (float*) (ws + off); off += (size_t)BATCH * NH * DHD * 4;

    const unsigned total = ROWS * DM;                 // 12,582,912
    copy_f32<<<total / 256, 256, 0, stream>>>(x, xo);

    const dim3 gN768(DM / 128, ROWS / 128);    // (6,128)
    const dim3 gN3072(HID / 128, ROWS / 128);  // (24,128)

    for (int i = 0; i < NDEPTH; ++i) {
        ln_kernel<<<ROWS, 256, 0, stream>>>(xo, ln1_g + i * DM, ln1_b + i * DM, hbf);
        gemm_bf16_wmma<0><<<gN768, 256, 0, stream>>>(
            hbf, qkv_w + (size_t)i * DM * DM, nullptr, nullptr,
            wbuf, nullptr, ROWS, DM, DM);
        att_nrm2<<<BATCH * NH, 256, 0, stream>>>(wbuf, nrm2);
        att_softmax<<<BATCH * NH, 256, 0, stream>>>(wbuf, nrm2, temp + i * NH, Pi, Tsum);
        att_dots<<<BATCH * NH, 256, 0, stream>>>(wbuf, Pi, Tsum, attn);
        att_out<<<total / 256, 256, 0, stream>>>(wbuf, Pi, attn, aout);
        gemm_bf16_wmma<1><<<gN768, 256, 0, stream>>>(
            aout, out_w + (size_t)i * DM * DM, out_b + i * DM, xo,
            xo, nullptr, ROWS, DM, DM);
        ln_kernel<<<ROWS, 256, 0, stream>>>(xo, ln2_g + i * DM, ln2_b + i * DM, hbf);
        gemm_bf16_wmma<2><<<gN3072, 256, 0, stream>>>(
            hbf, fc1_w + (size_t)i * DM * HID, fc1_b + i * HID, nullptr,
            nullptr, midbf, ROWS, HID, DM);
        gemm_bf16_wmma<1><<<gN768, 256, 0, stream>>>(
            midbf, fc2_w + (size_t)i * HID * DM, fc2_b + i * DM, xo,
            xo, nullptr, ROWS, DM, HID);
    }
}
